// TTULanguageModel_32882269618759
// MI455X (gfx1250) — matrix operand order repaired and checked by reading, not compiled, not run on hardware
//
#include <hip/hip_runtime.h>
#include <cstdint>
#include <cstddef>

typedef __attribute__((ext_vector_type(16))) __bf16 v16bf;
typedef __attribute__((ext_vector_type(8)))  float  v8f;

struct alignas(16) U4 { unsigned x, y, z, w; };
struct alignas(16) F4 { float x, y, z, w; };
struct alignas(8)  U2 { unsigned x, y; };

#define TILE_M 128
#define TILE_N 128
#define TILE_K 32

__device__ __forceinline__ unsigned short f2bf(float f) {
    unsigned u = __float_as_uint(f);
    unsigned r = u + 0x7FFFu + ((u >> 16) & 1u);   // round-to-nearest-even
    return (unsigned short)(r >> 16);
}

// ---- CDNA5 async global->LDS copy (ASYNCcnt path) ----------------------
// INST_OFFSET is added to BOTH the LDS and the global address (ISA 08 §4.4).
__device__ __forceinline__ void async_ld16(unsigned lds, unsigned long long ga) {
    asm volatile("global_load_async_to_lds_b128 %0, %1, off"
                 :: "v"(lds), "v"(ga) : "memory");
}
__device__ __forceinline__ void async_ld16_off16(unsigned lds, unsigned long long ga) {
    asm volatile("global_load_async_to_lds_b128 %0, %1, off offset:16"
                 :: "v"(lds), "v"(ga) : "memory");
}
__device__ __forceinline__ void wait_async0() {
    asm volatile("s_wait_asynccnt 0x0" ::: "memory");
}

// ---------------------------------------------------------------------------
// 1) impulses[s, b] = dot(hidden[b,s,:], W_imp[0,:]) + b_imp[0]
//    (the scan only consumes impulse component d=0)
// ---------------------------------------------------------------------------
__global__ __launch_bounds__(256)
void impulse_kernel(const float* __restrict__ hidden,
                    const float* __restrict__ Wimp,
                    const float* __restrict__ bimp,
                    float* __restrict__ imp,          // [S][B]
                    int H, int S, int B)
{
    int m    = blockIdx.x * 8 + (threadIdx.x >> 5);  // row index (b*S + s)
    int lane = threadIdx.x & 31;
    const float* h = hidden + (size_t)m * H;
    float a = 0.f;
    for (int k = lane; k < H; k += 32) a += h[k] * Wimp[k];
    #pragma unroll
    for (int off = 16; off > 0; off >>= 1) a += __shfl_xor(a, off, 32);
    if (lane == 0) {
        int s = m % S, b = m / S;
        imp[s * B + b] = a + bimp[0];
    }
}

// ---------------------------------------------------------------------------
// 2) sequential TTU scan (lane = batch), 1024 steps, trivial cost
// ---------------------------------------------------------------------------
__global__ void scan_kernel(const float* __restrict__ ttu_state,
                            const float* __restrict__ p,      // 7 params
                            const float* __restrict__ imp,    // [S][B]
                            float* __restrict__ states,       // [S][B][3]
                            float* __restrict__ finalState,   // [B][3]
                            int S, int B)
{
    int b = threadIdx.x;
    if (b >= B) return;
    const float g = p[0], ac = p[1], bc = p[2], de = p[3],
                ad = p[4], ep = p[5], am = p[6];
    float c = ttu_state[b * 3 + 0];
    float d = ttu_state[b * 3 + 1];
    float m = ttu_state[b * 3 + 2];
    for (int s = 0; s < S; ++s) {
        float im = imp[s * B + b];
        float dc = g * im - ac * c - bc * c * d;
        float dd = de * c * c - ad * d;
        float dm = ep * d - am * m;
        c += 0.1f * dc;
        d += 0.1f * dd;
        m += 0.1f * dm;
        d = (fabsf(d) > 10.0f) ? 0.5f * d : d;
        float* st = states + (size_t)(s * B + b) * 3;
        st[0] = c; st[1] = d; st[2] = m;
    }
    finalState[b * 3 + 0] = c;
    finalState[b * 3 + 1] = d;
    finalState[b * 3 + 2] = m;
}

// ---------------------------------------------------------------------------
// 3) fp32 -> bf16 conversion (vectorized, grid-stride)
// ---------------------------------------------------------------------------
__global__ __launch_bounds__(256)
void cvt_bf16_kernel(const float* __restrict__ src,
                     unsigned short* __restrict__ dst, long long n4)
{
    long long i      = (long long)blockIdx.x * blockDim.x + threadIdx.x;
    long long stride = (long long)gridDim.x * blockDim.x;
    const F4* s4 = (const F4*)src;
    U2* d2 = (U2*)dst;
    for (; i < n4; i += stride) {
        F4 v = s4[i];
        U2 o;
        o.x = ((unsigned)f2bf(v.y) << 16) | f2bf(v.x);
        o.y = ((unsigned)f2bf(v.w) << 16) | f2bf(v.z);
        d2[i] = o;
    }
}

// ---------------------------------------------------------------------------
// 4) logits[m, n] = sum_k Abf[m,k]*Wbf[n,k]  +  0.1 * states[m,:]·Wmod[n,:]
//    bf16 WMMA 16x16x32, fp32 accumulate. 128x128 block tile, 8 waves,
//    each wave = 64(M) x 32(N) = 4x2 WMMA accumulators.
//    Double-buffered LDS filled via GLOBAL_LOAD_ASYNC_TO_LDS_B128,
//    pre-swizzled into contiguous bf16 fragment K order:
//      A lane L (row M=L%16): K(i) = i + (L>=16?16:0)
//      B lane L (col N=L%16): K(i) = i + (L>=16?16:0)
// ---------------------------------------------------------------------------
__global__ __launch_bounds__(256)
void ttu_gemm_kernel(const unsigned short* __restrict__ Abf,   // [M][K] bf16
                     const unsigned short* __restrict__ Wbf,   // [V][K] bf16
                     const float* __restrict__ states,         // [S][B][3]
                     const float* __restrict__ Wmod,           // [V][3]
                     float* __restrict__ out,                  // [M][V]
                     int M, int K, int V, int S, int B, int mTiles)
{
    __shared__ U4 sA[2][8][32][2];       // 2 x 8 KB (8 M-subtiles of 16x32)
    __shared__ U4 sB[2][8][32][2];       // 2 x 8 KB (8 N-subtiles of 32x16)
    __shared__ float sSt[TILE_M][3];     // states for this M tile

    const int tid  = threadIdx.x;
    const int lane = tid & 31;
    const int wave = tid >> 5;
    const int wm = wave >> 2;            // 0..1  (M half of block tile)
    const int wn = wave & 3;             // 0..3  (N quarter)

    const int mt = blockIdx.x % mTiles;  // M fastest -> 16 adjacent blocks
    const int nt = blockIdx.x / mTiles;  //   share the same W columns (L2 hit)
    const int mBase = mt * TILE_M;
    const int nBase = nt * TILE_N;

    if (tid < TILE_M) {
        int m = mBase + tid;
        int s = m % S, b = m / S;
        const float* st = states + (size_t)(s * B + b) * 3;
        sSt[tid][0] = st[0]; sSt[tid][1] = st[1]; sSt[tid][2] = st[2];
    }

    // ---- per-thread async-copy assignments (computed once) ----
    // A: 512 16B chunks/tile, 2 per thread: chunk c -> (m16, L, half)
    unsigned long long aAddr[2];
    unsigned ldsA[2];
    #pragma unroll
    for (int cc = 0; cc < 2; ++cc) {
        int c    = tid + cc * 256;
        int m16  = c >> 6;
        int L    = (c & 63) >> 1;
        int half = c & 1;
        int mr   = L & 15;
        int Lhi  = L >> 4;
        int kst  = Lhi * 16 + half * 8;  // swizzled K start for this chunk
        aAddr[cc] = (unsigned long long)(size_t)
                    (Abf + (size_t)(mBase + m16 * 16 + mr) * K + kst);
        ldsA[cc]  = (unsigned)(size_t)&sA[0][m16][L][half];
    }
    // B: 32B per thread, contiguous in K: (n16, lane)
    const int nSub = tid >> 5;
    const int bL   = tid & 31;
    const int bLhi = bL >> 4;
    int bn = nBase + nSub * 16 + (bL & 15);
    if (bn >= V) bn = V - 1;             // clamp: OOB columns masked at store
    unsigned long long bAddr = (unsigned long long)(size_t)
                               (Wbf + (size_t)bn * K + bLhi * 16);
    const unsigned ldsB = (unsigned)(size_t)&sB[0][nSub][bL][0];

    const int nk = K / TILE_K;

    // prologue: fill buffer 0
    async_ld16(ldsA[0], aAddr[0]);
    async_ld16(ldsA[1], aAddr[1]);
    async_ld16(ldsB, bAddr);
    async_ld16_off16(ldsB, bAddr);       // offset:16 applies to LDS + global

    v8f acc[4][2] = {};

    for (int i = 0; i < nk; ++i) {
        const int buf = i & 1;
        wait_async0();                   // my loads for sX[buf] complete
        __syncthreads();                 // everyone's complete; prev buf free

        if (i + 1 < nk) {                // kick off next tile into other buf
            const unsigned boff = (unsigned)((i + 1) & 1) * 8192u;
            aAddr[0] += TILE_K * 2;
            aAddr[1] += TILE_K * 2;
            bAddr    += TILE_K * 2;
            async_ld16(ldsA[0] + boff, aAddr[0]);
            async_ld16(ldsA[1] + boff, aAddr[1]);
            async_ld16(ldsB + boff, bAddr);
            async_ld16_off16(ldsB + boff, bAddr);
        }

        v16bf af[4], bfr[2];
        #pragma unroll
        for (int j = 0; j < 4; ++j) {
            union { U4 q[2]; v16bf v; } u;
            u.q[0] = sA[buf][wm * 4 + j][lane][0];
            u.q[1] = sA[buf][wm * 4 + j][lane][1];
            af[j] = u.v;
        }
        #pragma unroll
        for (int jj = 0; jj < 2; ++jj) {
            union { U4 q[2]; v16bf v; } u;
            u.q[0] = sB[buf][wn * 2 + jj][lane][0];
            u.q[1] = sB[buf][wn * 2 + jj][lane][1];
            bfr[jj] = u.v;
        }
        #pragma unroll
        for (int j = 0; j < 4; ++j)
            #pragma unroll
            for (int jj = 0; jj < 2; ++jj)
                acc[j][jj] = __builtin_amdgcn_wmma_f32_16x16x32_bf16(
                    false, af[j], false, bfr[jj], (short)0, acc[j][jj],
                    false, false);
    }

    // ---- epilogue: add 0.1*states·Wmod and stream out (non-temporal) ----
    const int nlo  = lane & 15;
    const int Lhi8 = (lane >> 4) * 8;    // C/D layout: rows r / r+8 split
    #pragma unroll
    for (int jj = 0; jj < 2; ++jj) {
        int n = nBase + (wn * 2 + jj) * 16 + nlo;
        if (n >= V) continue;
        const float* wm3 = Wmod + (size_t)n * 3;
        float w0 = wm3[0], w1 = wm3[1], w2 = wm3[2];
        #pragma unroll
        for (int j = 0; j < 4; ++j) {
            int mrel0 = wm * 64 + j * 16 + Lhi8;
            #pragma unroll
            for (int r = 0; r < 8; ++r) {
                int mloc = mrel0 + r;
                float mod = 0.1f * (sSt[mloc][0] * w0 +
                                    sSt[mloc][1] * w1 +
                                    sSt[mloc][2] * w2);
                float val = acc[j][jj][r] + mod;
                __builtin_nontemporal_store(val, out + (size_t)(mBase + mloc) * V + n);
            }
        }
    }
}

// ---------------------------------------------------------------------------
extern "C" void kernel_launch(void* const* d_in, const int* in_sizes, int n_in,
                              void* d_out, int out_size, void* d_ws, size_t ws_size,
                              hipStream_t stream)
{
    const float* hidden     = (const float*)d_in[0];   // [B][S][H]
    const float* ttu_state  = (const float*)d_in[1];   // [B][3]
    const float* W_imp      = (const float*)d_in[2];   // [3][H]
    const float* b_imp      = (const float*)d_in[3];   // [3]
    const float* W_lm       = (const float*)d_in[4];   // [V][H]
    const float* W_mod      = (const float*)d_in[5];   // [V][3]
    const float* ttu_params = (const float*)d_in[6];   // [7]

    const int H = in_sizes[2] / 3;
    const int B = in_sizes[1] / 3;
    const int V = in_sizes[4] / H;
    const int S = in_sizes[0] / (B * H);
    const int M = B * S;                               // 2048

    // workspace layout
    char* ws = (char*)d_ws;
    float* imp    = (float*)ws;                        // M floats [S][B]
    float* states = imp + M;                           // M*3 floats [S][B][3]
    size_t off = (((size_t)M * 4 + (size_t)M * 3 * 4) + 255) & ~(size_t)255;
    unsigned short* Abf = (unsigned short*)(ws + off); // [M][H] bf16
    off += (size_t)M * H * 2;
    unsigned short* Wbf = (unsigned short*)(ws + off); // [V][H] bf16 (~103MB, L2-resident)

    float* logits     = (float*)d_out;                 // [M][V]
    float* finalState = logits + (size_t)M * V;        // [B][3]

    // 1) impulses (only component d=0 feeds the scan)
    impulse_kernel<<<M / 8, 256, 0, stream>>>(hidden, W_imp, b_imp, imp, H, S, B);

    // 2) sequential scan
    scan_kernel<<<1, 32, 0, stream>>>(ttu_state, ttu_params, imp, states,
                                      finalState, S, B);

    // 3) fp32 -> bf16 for GEMM operands
    cvt_bf16_kernel<<<2048, 256, 0, stream>>>(hidden, Abf, (long long)M * H / 4);
    cvt_bf16_kernel<<<8192, 256, 0, stream>>>(W_lm, Wbf, (long long)V * H / 4);

    // 4) fused GEMM + rank-3 modulation epilogue
    const int mTiles = M / TILE_M;
    const int nTiles = (V + TILE_N - 1) / TILE_N;
    ttu_gemm_kernel<<<mTiles * nTiles, 256, 0, stream>>>(
        Abf, Wbf, states, W_mod, logits, M, H, V, S, B, mTiles);
}
